// Attention_53017076301881
// MI455X (gfx1250) — matrix operand order repaired and checked by reading, not compiled, not run on hardware
//
#include <hip/hip_runtime.h>
#include <hip/hip_bf16.h>

// ---------------------------------------------------------------------------
// RoPE multi-head attention for MI455X (gfx1250), wave32 + WMMA.
// B=2, L=2048, HIDDEN=1024, HEADS=16, HEAD_DIM=64.
// Compute-bound (~1000 FLOP/byte vs ~10 FLOP/byte HBM balance at 23.3 TB/s):
// all matrix math on v_wmma_f32_16x16x32_f16, f32 accumulate.
// Data movement: double-buffered LDS tiles filled with gfx1250 async
// global->LDS copies (ASYNCcnt) when the builtin is available.
// ---------------------------------------------------------------------------

#define HID   1024
#define HEADS 16
#define HDIM  64
#define SEQ   2048
#define BATCH 2
#define MTOT  (BATCH * SEQ)   // 4096 rows

typedef _Float16 half_t;
typedef _Float16 v8h  __attribute__((ext_vector_type(8)));
typedef _Float16 v16h __attribute__((ext_vector_type(16)));
typedef float    v8f  __attribute__((ext_vector_type(8)));

#if defined(__AMDGCN__) && __has_builtin(__builtin_amdgcn_global_load_async_to_lds_b128)
#define HAVE_ASYNC 1
#else
#define HAVE_ASYNC 0
#endif
#if defined(__AMDGCN__) && __has_builtin(__builtin_amdgcn_s_wait_asynccnt)
#define ASYNC_WAIT(n) __builtin_amdgcn_s_wait_asynccnt(n)
#else
#define ASYNC_WAIT(n)
#endif

static __device__ __forceinline__ v16h cat8(v8h lo, v8h hi) {
  return __builtin_shufflevector(lo, hi, 0,1,2,3,4,5,6,7,8,9,10,11,12,13,14,15);
}
static __device__ __forceinline__ v8f wmma_f16(v16h a, v16h b, v8f c) {
  // (neg_a, A, neg_b, B, c_mod, C, reuse_a, reuse_b)
  return __builtin_amdgcn_wmma_f32_16x16x32_f16(false, a, false, b, (short)0, c,
                                                false, false);
}

// one 16-byte global->LDS copy (async path if available)
#if HAVE_ASYNC
typedef int v4i_t __attribute__((vector_size(16)));
static __device__ __forceinline__ void copy_b128(const half_t* g, half_t* l) {
  __builtin_amdgcn_global_load_async_to_lds_b128(
      (__attribute__((address_space(1))) v4i_t*)g,
      (__attribute__((address_space(3))) v4i_t*)l, 0, 0);
}
#else
static __device__ __forceinline__ void copy_b128(const half_t* g, half_t* l) {
  *(v8h*)l = *(const v8h*)g;
}
#endif

// ---------------------------------------------------------------------------
// f32 -> f16 conversion, 4 elements/thread
// ---------------------------------------------------------------------------
__global__ void cvt_f32_f16_kernel(const float* __restrict__ in,
                                   half_t* __restrict__ out, int n4) {
  int i = blockIdx.x * blockDim.x + threadIdx.x;
  if (i < n4) {
    float4 v = ((const float4*)in)[i];
    out[i * 4 + 0] = (half_t)v.x;
    out[i * 4 + 1] = (half_t)v.y;
    out[i * 4 + 2] = (half_t)v.z;
    out[i * 4 + 3] = (half_t)v.w;
  }
}

// ---------------------------------------------------------------------------
// RoPE tables: cos/sin [SEQ][HDIM], emb = concat(freqs,freqs)
// ---------------------------------------------------------------------------
__global__ void rope_tables_kernel(float* __restrict__ cosT,
                                   float* __restrict__ sinT) {
  int i = blockIdx.x * blockDim.x + threadIdx.x;  // SEQ*HDIM threads
  int t = i >> 6;
  int d = i & 63;
  int j = d & 31;
  float freq = __expf(-0.287823136624259f * (float)j);  // 10000^(-j/32)
  float v = (float)t * freq;
  float s, c;
  __sincosf(v, &s, &c);
  cosT[i] = c;
  sinT[i] = s;
}

// ---------------------------------------------------------------------------
// Shared GEMM mainloop: 128x128 block tile, K-step 64, double-buffered LDS.
// A from `am` (row-major [*,HID]), B columns from rows of `wm` ([N,HID]).
// 8 waves; each wave: 16 rows x 128 cols = 8 f32 accumulators.
// ---------------------------------------------------------------------------
static __device__ __forceinline__ void gemm_mainloop(
    const half_t* __restrict__ am, const half_t* __restrict__ wm,
    half_t* As,  // [2][128*72]
    half_t* Bs,  // [2][128*72]
    int m0, int n0, int tid, v8f acc[8]) {
  const int wave = tid >> 5, lane = tid & 31;
  const int col16 = lane & 15, halfsel = lane >> 4;
  // loader: 128 rows x 8 segs of 8 halves per matrix; 4 segs per thread
  const int lrow = tid >> 1;                 // 0..127
  const int lseg0 = (tid & 1) * 4;           // 0 or 4 -> segs lseg0..lseg0+3

  // prologue: stage tile 0
#pragma unroll
  for (int s = 0; s < 4; ++s) {
    int seg = lseg0 + s;
    copy_b128(&am[(size_t)(m0 + lrow) * HID + seg * 8], &As[lrow * 72 + seg * 8]);
    copy_b128(&wm[(size_t)(n0 + lrow) * HID + seg * 8], &Bs[lrow * 72 + seg * 8]);
  }

  const int ar = wave * 16 + col16;
  const int kb = halfsel * 16;   // A frag: K 0..15 | 16..31
  const int kb2 = halfsel * 16;  // B frag: K 0..15 | 16..31

  int p = 0;
  for (int kt = 0; kt < HID / 64; ++kt) {
    const int q = p ^ 1;
    if (kt + 1 < HID / 64) {  // stage next tile into the other buffer
      const int k0n = (kt + 1) * 64;
#pragma unroll
      for (int s = 0; s < 4; ++s) {
        int seg = lseg0 + s;
        copy_b128(&am[(size_t)(m0 + lrow) * HID + k0n + seg * 8],
                  &As[q * (128 * 72) + lrow * 72 + seg * 8]);
        copy_b128(&wm[(size_t)(n0 + lrow) * HID + k0n + seg * 8],
                  &Bs[q * (128 * 72) + lrow * 72 + seg * 8]);
      }
      ASYNC_WAIT(8);  // current tile's 8 async copies have retired (in-order)
    } else {
      ASYNC_WAIT(0);
    }
    __syncthreads();  // everyone's tile data visible

    const half_t* Ap = &As[p * (128 * 72)];
    const half_t* Bp = &Bs[p * (128 * 72)];
#pragma unroll
    for (int c = 0; c < 2; ++c) {  // two K=32 chunks
      v16h a = cat8(*(const v8h*)&Ap[ar * 72 + c * 32 + kb],
                    *(const v8h*)&Ap[ar * 72 + c * 32 + kb + 8]);
#pragma unroll
      for (int fg = 0; fg < 2; ++fg) {  // groups of 4: batch loads, then WMMAs
        v16h b0, b1, b2, b3;
        {
          int br = (fg * 4 + 0) * 16 + col16;
          b0 = cat8(*(const v8h*)&Bp[br * 72 + c * 32 + kb2],
                    *(const v8h*)&Bp[br * 72 + c * 32 + kb2 + 8]);
        }
        {
          int br = (fg * 4 + 1) * 16 + col16;
          b1 = cat8(*(const v8h*)&Bp[br * 72 + c * 32 + kb2],
                    *(const v8h*)&Bp[br * 72 + c * 32 + kb2 + 8]);
        }
        {
          int br = (fg * 4 + 2) * 16 + col16;
          b2 = cat8(*(const v8h*)&Bp[br * 72 + c * 32 + kb2],
                    *(const v8h*)&Bp[br * 72 + c * 32 + kb2 + 8]);
        }
        {
          int br = (fg * 4 + 3) * 16 + col16;
          b3 = cat8(*(const v8h*)&Bp[br * 72 + c * 32 + kb2],
                    *(const v8h*)&Bp[br * 72 + c * 32 + kb2 + 8]);
        }
        acc[fg * 4 + 0] = wmma_f16(a, b0, acc[fg * 4 + 0]);
        acc[fg * 4 + 1] = wmma_f16(a, b1, acc[fg * 4 + 1]);
        acc[fg * 4 + 2] = wmma_f16(a, b2, acc[fg * 4 + 2]);
        acc[fg * 4 + 3] = wmma_f16(a, b3, acc[fg * 4 + 3]);
      }
    }
    __syncthreads();  // all waves done reading buf p before it is refilled
    p = q;
  }
}

// ---------------------------------------------------------------------------
// Fused QKV projection + RoPE epilogue, scatter to [B,HEADS,SEQ,HDIM] f16.
// RoPE partner q[d +- 32] lives in the same lane, fragment f +- 2.
// ---------------------------------------------------------------------------
__global__ __launch_bounds__(256) void gemm_qkv_kernel(
    const half_t* __restrict__ xh,
    const half_t* __restrict__ wq, const half_t* __restrict__ wk,
    const half_t* __restrict__ wv,
    half_t* __restrict__ qo, half_t* __restrict__ ko, half_t* __restrict__ vo,
    const float* __restrict__ cosT, const float* __restrict__ sinT) {
  const int which = blockIdx.z;
  const half_t* w = (which == 0) ? wq : ((which == 1) ? wk : wv);
  half_t* outp = (which == 0) ? qo : ((which == 1) ? ko : vo);
  const bool rope = (which < 2);

  __shared__ half_t As[2 * 128 * 72];
  __shared__ half_t Bs[2 * 128 * 72];

  const int tid = threadIdx.x;
  const int wave = tid >> 5, lane = tid & 31;
  const int col16 = lane & 15, halfsel = lane >> 4;
  const int m0 = blockIdx.x * 128, n0 = blockIdx.y * 128;

  v8f acc[8] = {};
  gemm_mainloop(xh, w, As, Bs, m0, n0, tid, acc);

#pragma unroll
  for (int f = 0; f < 8; ++f) {
    const int fq = f & 3;  // position within the 64-wide head
    const int head = (n0 >> 6) + (f >> 2);
    const int dpos = col16 + 16 * fq;
#pragma unroll
    for (int i = 0; i < 8; ++i) {
      int r = i + halfsel * 8;
      int m = m0 + wave * 16 + r;
      int bb = m >> 11;  // batch (SEQ=2048)
      int t = m & (SEQ - 1);
      float val = acc[f][i];
      float res = val;
      if (rope) {
        float partner = (fq < 2) ? acc[f + 2][i] : acc[f - 2][i];
        float c = cosT[t * HDIM + dpos];
        float s = sinT[t * HDIM + dpos];
        res = (fq < 2) ? (val * c - partner * s) : (val * c + partner * s);
      }
      outp[(((size_t)bb * HEADS + head) * SEQ + t) * HDIM + dpos] = (half_t)res;
    }
  }
}

// ---------------------------------------------------------------------------
// Flash attention: grid (SEQ/128, B*HEADS), 8 waves x 16 query rows.
// K tile row-major (async-copied); V tile transposed [d][j] in LDS so the
// P.V B-fragment is a contiguous read. Online softmax; P converts C-layout
// -> A-layout through per-wave LDS scratch.
// ---------------------------------------------------------------------------
__global__ __launch_bounds__(256) void flash_attn_kernel(
    const half_t* __restrict__ qh, const half_t* __restrict__ kh,
    const half_t* __restrict__ vh, half_t* __restrict__ ah) {
  __shared__ half_t Kt[64 * 72];
  __shared__ half_t Vt[64 * 72];
  __shared__ half_t Ps[8 * 16 * 72];

  const int tid = threadIdx.x;
  const int wave = tid >> 5, lane = tid & 31;
  const int col16 = lane & 15, halfsel = lane >> 4;
  const int bh = blockIdx.y;
  const size_t base = (size_t)bh * SEQ * HDIM;
  const half_t* qp = qh + base;
  const half_t* kp = kh + base;
  const half_t* vp = vh + base;
  const int q0 = blockIdx.x * 128;
  const int qrow = q0 + wave * 16 + col16;
  const int kb = halfsel * 16;

  v16h qf[2];
#pragma unroll
  for (int c = 0; c < 2; ++c)
    qf[c] = cat8(*(const v8h*)&qp[(size_t)qrow * HDIM + c * 32 + kb],
                 *(const v8h*)&qp[(size_t)qrow * HDIM + c * 32 + kb + 8]);

  v8f o[4] = {};
  float mrow[8], lrow[8];
#pragma unroll
  for (int i = 0; i < 8; ++i) { mrow[i] = -1e30f; lrow[i] = 0.f; }

  const int pbase = wave * 16 * 72;
  // loader: 64 rows x 8 segs; 2 (row,seg) pairs per thread
  const int r0 = tid >> 3, s0 = tid & 7;          // first pair
  const int r1 = (tid + 256) >> 3, s1 = tid & 7;  // second pair

  for (int j0 = 0; j0 < SEQ; j0 += 64) {
    __syncthreads();
    // K tile: async straight to LDS
    copy_b128(&kp[(size_t)(j0 + r0) * HDIM + s0 * 8], &Kt[r0 * 72 + s0 * 8]);
    copy_b128(&kp[(size_t)(j0 + r1) * HDIM + s1 * 8], &Kt[r1 * 72 + s1 * 8]);
    // V tile: transpose on the way into LDS
    {
      v8h vv0 = *(const v8h*)&vp[(size_t)(j0 + r0) * HDIM + s0 * 8];
      v8h vv1 = *(const v8h*)&vp[(size_t)(j0 + r1) * HDIM + s1 * 8];
#pragma unroll
      for (int e = 0; e < 8; ++e) Vt[(s0 * 8 + e) * 72 + r0] = vv0[e];
#pragma unroll
      for (int e = 0; e < 8; ++e) Vt[(s1 * 8 + e) * 72 + r1] = vv1[e];
    }
    ASYNC_WAIT(0);
    __syncthreads();

    // S = Q . K^T : 16x64, 8 WMMAs
    v8f sf[4] = {};
#pragma unroll
    for (int c = 0; c < 2; ++c) {
      const int kb2 = halfsel * 16 + c * 32;
      v16h b0, b1, b2, b3;
      b0 = cat8(*(const v8h*)&Kt[(0 * 16 + col16) * 72 + kb2],
                *(const v8h*)&Kt[(0 * 16 + col16) * 72 + kb2 + 8]);
      b1 = cat8(*(const v8h*)&Kt[(1 * 16 + col16) * 72 + kb2],
                *(const v8h*)&Kt[(1 * 16 + col16) * 72 + kb2 + 8]);
      b2 = cat8(*(const v8h*)&Kt[(2 * 16 + col16) * 72 + kb2],
                *(const v8h*)&Kt[(2 * 16 + col16) * 72 + kb2 + 8]);
      b3 = cat8(*(const v8h*)&Kt[(3 * 16 + col16) * 72 + kb2],
                *(const v8h*)&Kt[(3 * 16 + col16) * 72 + kb2 + 8]);
      sf[0] = wmma_f16(qf[c], b0, sf[0]);
      sf[1] = wmma_f16(qf[c], b1, sf[1]);
      sf[2] = wmma_f16(qf[c], b2, sf[2]);
      sf[3] = wmma_f16(qf[c], b3, sf[3]);
    }
    const float sc = 0.125f;  // 1/sqrt(64)
#pragma unroll
    for (int fj = 0; fj < 4; ++fj) sf[fj] *= sc;

    // online softmax per row (rows replicated over each half-wave)
#pragma unroll
    for (int i = 0; i < 8; ++i) {
      float rm = fmaxf(fmaxf(sf[0][i], sf[1][i]), fmaxf(sf[2][i], sf[3][i]));
      rm = fmaxf(rm, __shfl_xor(rm, 1, 32));
      rm = fmaxf(rm, __shfl_xor(rm, 2, 32));
      rm = fmaxf(rm, __shfl_xor(rm, 4, 32));
      rm = fmaxf(rm, __shfl_xor(rm, 8, 32));
      float mnew = fmaxf(mrow[i], rm);
      float corr = __expf(mrow[i] - mnew);
      mrow[i] = mnew;
      float rs = 0.f;
#pragma unroll
      for (int fj = 0; fj < 4; ++fj) {
        float pv = __expf(sf[fj][i] - mnew);
        sf[fj][i] = pv;
        rs += pv;
      }
      rs += __shfl_xor(rs, 1, 32);
      rs += __shfl_xor(rs, 2, 32);
      rs += __shfl_xor(rs, 4, 32);
      rs += __shfl_xor(rs, 8, 32);
      lrow[i] = lrow[i] * corr + rs;
#pragma unroll
      for (int fd = 0; fd < 4; ++fd) o[fd][i] *= corr;
      int r = i + halfsel * 8;
#pragma unroll
      for (int fj = 0; fj < 4; ++fj)
        Ps[pbase + r * 72 + col16 + 16 * fj] = (half_t)sf[fj][i];
    }

    // O += P . V : 8 WMMAs (per-wave LDS is in-order: stores precede loads)
#pragma unroll
    for (int c = 0; c < 2; ++c) {
      v16h pa = cat8(*(const v8h*)&Ps[pbase + col16 * 72 + c * 32 + kb],
                     *(const v8h*)&Ps[pbase + col16 * 72 + c * 32 + kb + 8]);
      const int kb2 = halfsel * 16 + c * 32;
      v16h b0, b1, b2, b3;
      b0 = cat8(*(const v8h*)&Vt[(0 * 16 + col16) * 72 + kb2],
                *(const v8h*)&Vt[(0 * 16 + col16) * 72 + kb2 + 8]);
      b1 = cat8(*(const v8h*)&Vt[(1 * 16 + col16) * 72 + kb2],
                *(const v8h*)&Vt[(1 * 16 + col16) * 72 + kb2 + 8]);
      b2 = cat8(*(const v8h*)&Vt[(2 * 16 + col16) * 72 + kb2],
                *(const v8h*)&Vt[(2 * 16 + col16) * 72 + kb2 + 8]);
      b3 = cat8(*(const v8h*)&Vt[(3 * 16 + col16) * 72 + kb2],
                *(const v8h*)&Vt[(3 * 16 + col16) * 72 + kb2 + 8]);
      o[0] = wmma_f16(pa, b0, o[0]);
      o[1] = wmma_f16(pa, b1, o[1]);
      o[2] = wmma_f16(pa, b2, o[2]);
      o[3] = wmma_f16(pa, b3, o[3]);
    }
  }

  // normalize + write [B, SEQ, HID] f16 (re-interleave heads for out proj)
  const int b = bh >> 4, h = bh & 15;
#pragma unroll
  for (int i = 0; i < 8; ++i) {
    int r = i + halfsel * 8;
    int t = q0 + wave * 16 + r;
    float inv = 1.f / lrow[i];
#pragma unroll
    for (int fd = 0; fd < 4; ++fd) {
      int d = col16 + 16 * fd;
      ah[((size_t)b * SEQ + t) * HID + h * HDIM + d] = (half_t)(o[fd][i] * inv);
    }
  }
}

// ---------------------------------------------------------------------------
// Output projection: d_out = attn_out @ Wo^T, f32 stores.
// ---------------------------------------------------------------------------
__global__ __launch_bounds__(256) void gemm_out_kernel(
    const half_t* __restrict__ ahh, const half_t* __restrict__ wo,
    float* __restrict__ outp) {
  __shared__ half_t As[2 * 128 * 72];
  __shared__ half_t Bs[2 * 128 * 72];
  const int tid = threadIdx.x;
  const int wave = tid >> 5, lane = tid & 31;
  const int col16 = lane & 15, halfsel = lane >> 4;
  const int m0 = blockIdx.x * 128, n0 = blockIdx.y * 128;

  v8f acc[8] = {};
  gemm_mainloop(ahh, wo, As, Bs, m0, n0, tid, acc);

#pragma unroll
  for (int f = 0; f < 8; ++f) {
#pragma unroll
    for (int i = 0; i < 8; ++i) {
      int r = i + halfsel * 8;
      int m = m0 + wave * 16 + r;
      outp[(size_t)m * HID + n0 + col16 + 16 * f] = acc[f][i];
    }
  }
}

// ---------------------------------------------------------------------------
// Host launcher
// ---------------------------------------------------------------------------
extern "C" void kernel_launch(void* const* d_in, const int* in_sizes, int n_in,
                              void* d_out, int out_size, void* d_ws,
                              size_t ws_size, hipStream_t stream) {
  const float* x  = (const float*)d_in[0];
  const float* Wq = (const float*)d_in[1];
  const float* Wk = (const float*)d_in[2];
  const float* Wv = (const float*)d_in[3];
  const float* Wo = (const float*)d_in[4];
  float* out = (float*)d_out;

  char* ws = (char*)d_ws;
  size_t off = 0;
  auto wsalloc = [&](size_t bytes) -> void* {
    void* p = ws + off;
    off += (bytes + 255) & ~(size_t)255;
    return p;
  };
  half_t* xh  = (half_t*)wsalloc((size_t)MTOT * HID * 2);   // x in f16
  half_t* wqh = (half_t*)wsalloc((size_t)HID * HID * 2);
  half_t* wkh = (half_t*)wsalloc((size_t)HID * HID * 2);
  half_t* wvh = (half_t*)wsalloc((size_t)HID * HID * 2);
  half_t* woh = (half_t*)wsalloc((size_t)HID * HID * 2);
  half_t* qh  = (half_t*)wsalloc((size_t)MTOT * HID * 2);   // [B,H,L,D]
  half_t* kh  = (half_t*)wsalloc((size_t)MTOT * HID * 2);   // [B,H,L,D]
  half_t* vh  = (half_t*)wsalloc((size_t)MTOT * HID * 2);   // [B,H,L,D]
  half_t* ah  = (half_t*)wsalloc((size_t)MTOT * HID * 2);   // [B,L,HID]
  float* cosT = (float*)wsalloc((size_t)SEQ * HDIM * 4);
  float* sinT = (float*)wsalloc((size_t)SEQ * HDIM * 4);

  // 1) precision conversion + RoPE tables
  cvt_f32_f16_kernel<<<(MTOT * HID / 4) / 256, 256, 0, stream>>>(
      x, xh, MTOT * HID / 4);
  cvt_f32_f16_kernel<<<(HID * HID / 4) / 256, 256, 0, stream>>>(
      Wq, wqh, HID * HID / 4);
  cvt_f32_f16_kernel<<<(HID * HID / 4) / 256, 256, 0, stream>>>(
      Wk, wkh, HID * HID / 4);
  cvt_f32_f16_kernel<<<(HID * HID / 4) / 256, 256, 0, stream>>>(
      Wv, wvh, HID * HID / 4);
  cvt_f32_f16_kernel<<<(HID * HID / 4) / 256, 256, 0, stream>>>(
      Wo, woh, HID * HID / 4);
  rope_tables_kernel<<<(SEQ * HDIM) / 256, 256, 0, stream>>>(cosT, sinT);

  // 2) fused QKV projection + RoPE
  gemm_qkv_kernel<<<dim3(MTOT / 128, HID / 128, 3), 256, 0, stream>>>(
      xh, wqh, wkh, wvh, qh, kh, vh, cosT, sinT);

  // 3) flash attention
  flash_attn_kernel<<<dim3(SEQ / 128, BATCH * HEADS), 256, 0, stream>>>(
      qh, kh, vh, ah);

  // 4) output projection
  gemm_out_kernel<<<dim3(MTOT / 128, HID / 128), 256, 0, stream>>>(ah, woh,
                                                                   out);
}
